// WPFormer_35785667510836
// MI455X (gfx1250) — matrix operand order repaired and checked by reading, not compiled, not run on hardware
//
#include <hip/hip_runtime.h>
#include <math.h>

typedef _Float16 f16;
typedef _Float16 v16h __attribute__((ext_vector_type(16)));
typedef float    v8f  __attribute__((ext_vector_type(8)));

#define DEV static __device__ __forceinline__

// ---- WMMA fragment helpers (CDNA5 16x16x32 f16 layouts, ISA 7.12.2) ----
// A (16x32 f16): lane L (row = L&15): halves[0..7] = K kg..kg+7, halves[8..15] = K kg+16..kg+23,
// kg = ktile*32 + (L>=16 ? 8 : 0)
DEV v16h frag_a(const f16* base, int row_stride, int rowbase, int ktile, int lane) {
    int row = rowbase + (lane & 15);
    int kg  = ktile * 32 + ((lane >> 4) << 3);
    const f16* p0 = base + (size_t)row * row_stride + kg;
    const f16* p1 = p0 + 16;
    v16h r;
#pragma unroll
    for (int i = 0; i < 8; ++i) { r[i] = p0[i]; r[8 + i] = p1[i]; }
    return r;
}

// B (32x16 f16) from W stored as [n][k] (i.e. Wt so out = x @ W^T):
// lane L (col n = L&15): halves[0..7] = W[n][kg..kg+7], halves[8..15] = W[n][kg+16..kg+23], kg = ktile*32 + (L>=16 ? 8 : 0)
DEV v16h frag_b(const f16* wh, int ntile, int ktile, int lane) {
    int n  = ntile * 16 + (lane & 15);
    int kg = ktile * 32 + ((lane >> 4) << 3);
    const f16* p = wh + n * 64 + kg;
    v16h r;
#pragma unroll
    for (int i = 0; i < 8; ++i) { r[i] = p[i]; r[8 + i] = p[16 + i]; }
    return r;
}

DEV v8f wmma16(v16h a, v16h b, v8f c) {
    return __builtin_amdgcn_wmma_f32_16x16x32_f16(false, a, false, b, (short)0, c, false, false);
}

DEV v8f vzero8() { v8f z = {0.f,0.f,0.f,0.f,0.f,0.f,0.f,0.f}; return z; }

// ---------------- constants ----------------
#define BB   8
#define DD   64
#define NQ   16
#define N1T  16384
#define N2T  4096
#define PP   16
#define EPSF 1e-5f

// ---------------- kernels ----------------
__global__ void k_prep_weights(const float* w_in, const float* pw, const float* mh,
                               const float* lw1, const float* lw2,
                               const float* g1, const float* b1, const float* m1, const float* v1,
                               const float* g2, const float* b2, const float* m2, const float* v2,
                               f16* lw1h, f16* lw2h, f16* wkh, f16* wvh, f16* pwh, f16* mwh,
                               float* bn1s, float* bn1b, float* bn2s, float* bn2b) {
    for (int i = threadIdx.x; i < 64 * 64; i += blockDim.x) {
        lw1h[i] = (f16)lw1[i];
        lw2h[i] = (f16)lw2[i];
        wkh[i]  = (f16)w_in[64 * 64 + i];
        wvh[i]  = (f16)w_in[128 * 64 + i];
        pwh[i]  = (f16)pw[i];
    }
    for (int i = threadIdx.x; i < 16 * 64; i += blockDim.x) mwh[i] = (f16)mh[i];
    for (int i = threadIdx.x; i < 64; i += blockDim.x) {
        float s1 = rsqrtf(v1[i] + EPSF);
        bn1s[i] = g1[i] * s1; bn1b[i] = b1[i] - m1[i] * g1[i] * s1;
        float s2 = rsqrtf(v2[i] + EPSF);
        bn2s[i] = g2[i] * s2; bn2b[i] = b2[i] - m2[i] * g2[i] * s2;
    }
}

// key[b][n][c] = memory[n][b][c] + pos[n][b][c]
__global__ void k_key(const float* mem, const float* pos, float* key) {
    int i = blockIdx.x * blockDim.x + threadIdx.x;
    if (i >= BB * N1T * DD) return;
    int c = i & 63, n = (i >> 6) & (N1T - 1), b = i >> 20;
    int src = (n * BB + b) * DD + c;
    key[i] = mem[src] + pos[src];
}

__global__ void k_query(const float* tgt, const float* qp, float* qry) {
    int i = blockIdx.x * blockDim.x + threadIdx.x;
    if (i >= BB * NQ * DD) return;
    int c = i & 63, q = (i >> 6) & 15, b = i >> 10;
    int src = (q * BB + b) * DD + c;
    qry[i] = tgt[src] + qp[src];
}

// Haar 2x2 stride-2: low/high [B,4096,64]; s_h = f16(high+low)
__global__ void k_wavelet(const float* key, float* high, float* low, f16* s_h) {
    int i = blockIdx.x * blockDim.x + threadIdx.x;
    if (i >= BB * N2T * DD) return;
    int c = i & 63, n2 = (i >> 6) & (N2T - 1), b = i >> 18;
    int oy = n2 >> 6, ox = n2 & 63;
    const float* kb = key + ((size_t)b << 20);
    int iy = oy * 2, ix = ox * 2;
    float v00 = kb[(iy * 128 + ix) * 64 + c];
    float v01 = kb[(iy * 128 + ix + 1) * 64 + c];
    float v10 = kb[((iy + 1) * 128 + ix) * 64 + c];
    float v11 = kb[((iy + 1) * 128 + ix + 1) * 64 + c];
    float ll = 0.5f * ( v00 + v01 + v10 + v11);
    float lh = 0.5f * (-v00 + v01 - v10 + v11);
    float hl = 0.5f * (-v00 - v01 + v10 + v11);
    float hh = 0.5f * ( v00 - v01 - v10 + v11);
    float hi = lh + hl + hh;
    high[i] = hi; low[i] = ll; s_h[i] = (f16)(hi + ll);
}

// global-branch of MSCW-1: pool over 4096 tokens -> 2-layer MLP -> glo[b][64]
__global__ void k_pool_glo(const float* high, const float* low,
                           const float* gw1, const float* gb1,
                           const float* gw2, const float* gb2, float* glo) {
    int b = blockIdx.x, j = threadIdx.x;
    __shared__ float pool[64], h1[64];
    const float* hp = high + ((size_t)b << 18);
    const float* lp = low  + ((size_t)b << 18);
    float s = 0.f;
    for (int n = 0; n < N2T; ++n) s += hp[n * 64 + j] + lp[n * 64 + j];
    pool[j] = s * (1.0f / (float)N2T);
    __syncthreads();
    float t = gb1[j];
    for (int k = 0; k < 64; ++k) t += pool[k] * gw1[j * 64 + k];
    h1[j] = fmaxf(t, 0.f);
    __syncthreads();
    float g = gb2[j];
    for (int k = 0; k < 64; ++k) g += h1[k] * gw2[j * 64 + k];
    glo[b * 64 + j] = g;
}

// MSCW-1 local MLP (two 64x64 layers via WMMA) + sigmoid gate + fre = wei*high+low (f16)
__global__ void __launch_bounds__(32) k_mscw_local(const f16* s_h, const f16* lw1h, const f16* lw2h,
                                                   const float* lb1, const float* lb2, const float* glo,
                                                   const float* high, const float* low, f16* fre_h) {
    int lane = threadIdx.x;
    int rowbase = blockIdx.x * 16;     // rows over [B*4096]
    int b = rowbase >> 12;
    __shared__ f16 tile[16 * 64];
    int col0 = lane & 15, rbase = (lane >> 4) << 3;
#pragma unroll
    for (int nt = 0; nt < 4; ++nt) {
        v8f acc = vzero8();
#pragma unroll
        for (int kt = 0; kt < 2; ++kt)
            acc = wmma16(frag_a(s_h, 64, rowbase, kt, lane), frag_b(lw1h, nt, kt, lane), acc);
        int col = nt * 16 + col0;
        float bias = lb1[col];
#pragma unroll
        for (int r = 0; r < 8; ++r)
            tile[(rbase + r) * 64 + col] = (f16)fmaxf(acc[r] + bias, 0.f);
    }
    __syncthreads();
#pragma unroll
    for (int nt = 0; nt < 4; ++nt) {
        v8f acc = vzero8();
#pragma unroll
        for (int kt = 0; kt < 2; ++kt)
            acc = wmma16(frag_a(tile, 64, 0, kt, lane), frag_b(lw2h, nt, kt, lane), acc);
        int col = nt * 16 + col0;
        float bias = lb2[col] + glo[b * 64 + col];
#pragma unroll
        for (int r = 0; r < 8; ++r) {
            size_t gi = (size_t)(rowbase + rbase + r) * 64 + col;
            float wei = 1.0f / (1.0f + expf(-(acc[r] + bias)));
            fre_h[gi] = (f16)(wei * high[gi] + low[gi]);
        }
    }
}

// depthwise 3x3 (pad 1) + BN1 + ReLU -> y1 f16 [B*16384][64]
__global__ void k_dwconv(const float* key, const float* dww,
                         const float* bn1s, const float* bn1b, f16* y1) {
    int i = blockIdx.x * blockDim.x + threadIdx.x;
    if (i >= BB * N1T * DD) return;
    int c = i & 63, n = (i >> 6) & (N1T - 1), b = i >> 20;
    int y = n >> 7, x = n & 127;
    const float* kb = key + ((size_t)b << 20);
    float s = 0.f;
#pragma unroll
    for (int dy = -1; dy <= 1; ++dy) {
        int yy = y + dy; if (yy < 0 || yy > 127) continue;
#pragma unroll
        for (int dx = -1; dx <= 1; ++dx) {
            int xx = x + dx; if (xx < 0 || xx > 127) continue;
            s += kb[(yy * 128 + xx) * 64 + c] * dww[c * 9 + (dy + 1) * 3 + (dx + 1)];
        }
    }
    s = s * bn1s[c] + bn1b[c];
    y1[i] = (f16)fmaxf(s, 0.f);
}

// pointwise 1x1 (WMMA) + BN2 + ReLU, fused mask-head featf @ mheads^T (WMMA) -> mlog [B*16384][16]
__global__ void __launch_bounds__(32) k_pointwise(const f16* y1, const f16* pwh, const f16* mwh,
                                                  const float* bn2s, const float* bn2b, float* mlog) {
    int lane = threadIdx.x;
    int rowbase = blockIdx.x * 16;      // rows over [B*16384]
    __shared__ f16 tile[16 * 64];
    int col0 = lane & 15, rbase = (lane >> 4) << 3;
#pragma unroll
    for (int nt = 0; nt < 4; ++nt) {
        v8f acc = vzero8();
#pragma unroll
        for (int kt = 0; kt < 2; ++kt)
            acc = wmma16(frag_a(y1, 64, rowbase, kt, lane), frag_b(pwh, nt, kt, lane), acc);
        int col = nt * 16 + col0;
        float sc = bn2s[col], sh = bn2b[col];
#pragma unroll
        for (int r = 0; r < 8; ++r)
            tile[(rbase + r) * 64 + col] = (f16)fmaxf(acc[r] * sc + sh, 0.f);
    }
    __syncthreads();
    v8f acc = vzero8();
#pragma unroll
    for (int kt = 0; kt < 2; ++kt)
        acc = wmma16(frag_a(tile, 64, 0, kt, lane), frag_b(mwh, 0, kt, lane), acc);
#pragma unroll
    for (int r = 0; r < 8; ++r)
        mlog[(size_t)(rowbase + rbase + r) * 16 + col0] = acc[r];
}

// softmax over tokens (16384) per (b,p): max & sum
__global__ void k_softstats(const float* mlog, float* rmax, float* rsum) {
    int bp = blockIdx.x, b = bp >> 4, p = bp & 15;
    int tid = threadIdx.x;
    __shared__ float red[256];
    const float* base = mlog + (size_t)b * N1T * 16 + p;
    float m = -1e30f;
    for (int n = tid; n < N1T; n += 256) m = fmaxf(m, base[(size_t)n * 16]);
    red[tid] = m; __syncthreads();
    for (int s = 128; s > 0; s >>= 1) { if (tid < s) red[tid] = fmaxf(red[tid], red[tid + s]); __syncthreads(); }
    float gmax = red[0]; __syncthreads();
    float sum = 0.f;
    for (int n = tid; n < N1T; n += 256) sum += expf(base[(size_t)n * 16] - gmax);
    red[tid] = sum; __syncthreads();
    for (int s = 128; s > 0; s >>= 1) { if (tid < s) red[tid] += red[tid + s]; __syncthreads(); }
    if (tid == 0) { rmax[bp] = gmax; rsum[bp] = red[0]; }
}

// protos[b,p,c] = sum_n softmax(mlog)[b,n,p] * key[b,n,c]
__global__ void k_protos(const float* mlog, const float* key, const float* rmax,
                         const float* rsum, float* protos) {
    int bp = blockIdx.x, b = bp >> 4, p = bp & 15;
    int tid = threadIdx.x, c = tid & 63, sl = tid >> 6;
    __shared__ float red[256];
    float gmax = rmax[bp];
    const float* lg = mlog + (size_t)b * N1T * 16 + p;
    const float* kb = key + ((size_t)b << 20);
    float acc = 0.f;
    for (int n = sl * 4096; n < sl * 4096 + 4096; ++n)
        acc += expf(lg[(size_t)n * 16] - gmax) * kb[(size_t)n * 64 + c];
    red[tid] = acc; __syncthreads();
    if (sl == 0) {
        float t = red[c] + red[64 + c] + red[128 + c] + red[192 + c];
        protos[(size_t)bp * 64 + c] = t / rsum[bp];
    }
}

// q projection -> head-major qh[b][h][q][8]
__global__ void k_qproj(const float* qry, const float* w_in, const float* b_in, float* qh) {
    int bq = blockIdx.x, b = bq >> 4, q = bq & 15, j = threadIdx.x;
    const float* x = qry + (size_t)bq * 64;
    float s = b_in[j];
    for (int k = 0; k < 64; ++k) s += x[k] * w_in[j * 64 + k];
    int h = j >> 3, d = j & 7;
    qh[(((b * 8 + h) * 16 + q) * 8) + d] = s;
}

// K/V projections of fre via WMMA
__global__ void __launch_bounds__(32) k_kvproj(const f16* fre_h, const f16* wkh, const f16* wvh,
                                               const float* b_in, float* kp, float* vp) {
    int lane = threadIdx.x;
    int rowbase = blockIdx.x * 16;      // rows over [B*4096]
    int col0 = lane & 15, rbase = (lane >> 4) << 3;
#pragma unroll
    for (int nt = 0; nt < 4; ++nt) {
        v8f ak = vzero8(), av = vzero8();
#pragma unroll
        for (int kt = 0; kt < 2; ++kt) {
            v16h a = frag_a(fre_h, 64, rowbase, kt, lane);
            ak = wmma16(a, frag_b(wkh, nt, kt, lane), ak);
            av = wmma16(a, frag_b(wvh, nt, kt, lane), av);
        }
        int col = nt * 16 + col0;
        float bk = b_in[64 + col], bv = b_in[128 + col];
#pragma unroll
        for (int r = 0; r < 8; ++r) {
            size_t gi = (size_t)(rowbase + rbase + r) * 64 + col;
            kp[gi] = ak[r] + bk;
            vp[gi] = av[r] + bv;
        }
    }
}

// online-softmax attention per (b,h): 16 queries x 4096 keys, dk=8
__global__ void k_attn(const float* qh, const float* kp, const float* vp, float* o) {
    int b = blockIdx.x >> 3, h = blockIdx.x & 7;
    int tid = threadIdx.x, q = tid & 15, sl = tid >> 4;
    __shared__ float sm[256], sls[256], sac[256 * 8];
    float qv[8];
#pragma unroll
    for (int d = 0; d < 8; ++d) qv[d] = qh[(((b * 8 + h) * 16 + q) * 8) + d];
    const float scale = 0.3535533905932738f;    // 1/sqrt(8)
    float m = -1e30f, l = 0.f, acc[8] = {0, 0, 0, 0, 0, 0, 0, 0};
    const float* kpb = kp + (size_t)b * N2T * 64 + h * 8;
    const float* vpb = vp + (size_t)b * N2T * 64 + h * 8;
    for (int n = sl * 256; n < sl * 256 + 256; ++n) {
        const float* kk = kpb + (size_t)n * 64;
        float s = 0.f;
#pragma unroll
        for (int d = 0; d < 8; ++d) s += qv[d] * kk[d];
        s *= scale;
        float nm = fmaxf(m, s);
        float c1 = expf(m - nm), pwr = expf(s - nm);
        l = l * c1 + pwr;
        const float* vv = vpb + (size_t)n * 64;
#pragma unroll
        for (int d = 0; d < 8; ++d) acc[d] = acc[d] * c1 + pwr * vv[d];
        m = nm;
    }
    sm[tid] = m; sls[tid] = l;
#pragma unroll
    for (int d = 0; d < 8; ++d) sac[tid * 8 + d] = acc[d];
    __syncthreads();
    if (tid < 16) {
        float M = -1e30f, L = 0.f, A[8] = {0, 0, 0, 0, 0, 0, 0, 0};
        for (int s2 = 0; s2 < 16; ++s2) {
            int t = s2 * 16 + tid;
            float m2 = sm[t], nM = fmaxf(M, m2);
            float c1 = expf(M - nM), c2 = expf(m2 - nM);
            L = L * c1 + sls[t] * c2;
#pragma unroll
            for (int d = 0; d < 8; ++d) A[d] = A[d] * c1 + sac[t * 8 + d] * c2;
            M = nM;
        }
        float inv = 1.0f / L;
#pragma unroll
        for (int d = 0; d < 8; ++d)
            o[(size_t)(b * 16 + tid) * 64 + h * 8 + d] = A[d] * inv;
    }
}

// out-projection + residual + LN1 -> x1
__global__ void k_outln1(const float* o, const float* w_out, const float* b_out,
                         const float* qry, const float* g, const float* bbeta, float* x1) {
    int bq = blockIdx.x, j = threadIdx.x;
    __shared__ float red[64], stat[2];
    const float* xo = o + (size_t)bq * 64;
    float v = b_out[j];
    for (int k = 0; k < 64; ++k) v += xo[k] * w_out[j * 64 + k];
    v += qry[(size_t)bq * 64 + j];
    red[j] = v; __syncthreads();
    if (j == 0) { float s = 0; for (int k = 0; k < 64; ++k) s += red[k]; stat[0] = s / 64.f; }
    __syncthreads();
    float mu = stat[0], dv = v - mu;
    red[j] = dv * dv; __syncthreads();
    if (j == 0) { float s = 0; for (int k = 0; k < 64; ++k) s += red[k]; stat[1] = s / 64.f; }
    __syncthreads();
    x1[(size_t)bq * 64 + j] = dv * rsqrtf(stat[1] + EPSF) * g[j] + bbeta[j];
}

// MSCW-2 on (protos+query), gate, LN2, and final output transpose [Nq,B,C]
__global__ void k_final(const float* protos, const float* qry, const float* x1,
                        const float* lw1, const float* lb1, const float* lw2, const float* lb2,
                        const float* gw1, const float* gb1, const float* gw2, const float* gb2,
                        const float* g, const float* bbeta, float* out) {
    int b = blockIdx.x, j = threadIdx.x;
    __shared__ float pool[64], h1[64], red[64], stat[2];
    const float* pb = protos + (size_t)b * 16 * 64;
    const float* qb = qry + (size_t)b * 16 * 64;
    float s = 0.f;
    for (int q = 0; q < 16; ++q) s += pb[q * 64 + j] + qb[q * 64 + j];
    pool[j] = s / 16.f;
    __syncthreads();
    float t = gb1[j];
    for (int k = 0; k < 64; ++k) t += pool[k] * gw1[j * 64 + k];
    h1[j] = fmaxf(t, 0.f);
    __syncthreads();
    float gloj = gb2[j];
    for (int k = 0; k < 64; ++k) gloj += h1[k] * gw2[j * 64 + k];
    __syncthreads();
    for (int q = 0; q < 16; ++q) {
        float t1 = lb1[j];
        for (int k = 0; k < 64; ++k) t1 += (pb[q * 64 + k] + qb[q * 64 + k]) * lw1[j * 64 + k];
        h1[j] = fmaxf(t1, 0.f);
        __syncthreads();
        float loc = lb2[j];
        for (int k = 0; k < 64; ++k) loc += h1[k] * lw2[j * 64 + k];
        float a2 = 1.f / (1.f + expf(-(loc + gloj)));
        float qv = qb[q * 64 + j];
        float v = qv * a2 + qv;
        red[j] = v; __syncthreads();
        if (j == 0) { float ss = 0; for (int k = 0; k < 64; ++k) ss += red[k]; stat[0] = ss / 64.f; }
        __syncthreads();
        float mu = stat[0], dv = v - mu;
        red[j] = dv * dv; __syncthreads();
        if (j == 0) { float ss = 0; for (int k = 0; k < 64; ++k) ss += red[k]; stat[1] = ss / 64.f; }
        __syncthreads();
        float x2 = dv * rsqrtf(stat[1] + EPSF) * g[j] + bbeta[j];
        out[(size_t)(q * BB + b) * 64 + j] = x1[(size_t)(b * 16 + q) * 64 + j] + x2;
        __syncthreads();
    }
}

extern "C" void kernel_launch(void* const* d_in, const int* in_sizes, int n_in,
                              void* d_out, int out_size, void* d_ws, size_t ws_size,
                              hipStream_t stream) {
    (void)in_sizes; (void)n_in; (void)out_size; (void)ws_size;
    const float* tgt    = (const float*)d_in[0];
    const float* memory = (const float*)d_in[1];
    const float* pos    = (const float*)d_in[2];
    const float* qpos   = (const float*)d_in[3];
    const float* w_in   = (const float*)d_in[4];
    const float* b_in   = (const float*)d_in[5];
    const float* w_out  = (const float*)d_in[6];
    const float* b_out  = (const float*)d_in[7];
    const float* ln1g   = (const float*)d_in[8];
    const float* ln1b   = (const float*)d_in[9];
    const float* ln2g   = (const float*)d_in[10];
    const float* ln2b   = (const float*)d_in[11];
    const float* dww    = (const float*)d_in[12];
    const float* bn1g   = (const float*)d_in[13];
    const float* bn1bb  = (const float*)d_in[14];
    const float* bn1m   = (const float*)d_in[15];
    const float* bn1v   = (const float*)d_in[16];
    const float* pww    = (const float*)d_in[17];
    const float* bn2g   = (const float*)d_in[18];
    const float* bn2bb  = (const float*)d_in[19];
    const float* bn2m   = (const float*)d_in[20];
    const float* bn2v   = (const float*)d_in[21];
    const float* mhw    = (const float*)d_in[22];
    const float* m1lw1  = (const float*)d_in[23];
    const float* m1lb1  = (const float*)d_in[24];
    const float* m1lw2  = (const float*)d_in[25];
    const float* m1lb2  = (const float*)d_in[26];
    const float* m1gw1  = (const float*)d_in[27];
    const float* m1gb1  = (const float*)d_in[28];
    const float* m1gw2  = (const float*)d_in[29];
    const float* m1gb2  = (const float*)d_in[30];
    const float* m2lw1  = (const float*)d_in[31];
    const float* m2lb1  = (const float*)d_in[32];
    const float* m2lw2  = (const float*)d_in[33];
    const float* m2lb2  = (const float*)d_in[34];
    const float* m2gw1  = (const float*)d_in[35];
    const float* m2gb1  = (const float*)d_in[36];
    const float* m2gw2  = (const float*)d_in[37];
    const float* m2gb2  = (const float*)d_in[38];
    float* out = (float*)d_out;

    char* ws = (char*)d_ws;
    size_t off = 0;
    auto alloc = [&](size_t bytes) -> char* {
        char* p = ws + off;
        off = (off + bytes + 255) & ~(size_t)255;
        return p;
    };
    float* key    = (float*)alloc((size_t)BB * N1T * DD * 4);   // 33.5 MB
    f16*   s_h    = (f16*)  alloc((size_t)BB * N2T * DD * 2);
    float* high   = (float*)alloc((size_t)BB * N2T * DD * 4);
    float* low    = (float*)alloc((size_t)BB * N2T * DD * 4);
    float* glo1   = (float*)alloc((size_t)BB * DD * 4);
    f16*   fre_h  = (f16*)  alloc((size_t)BB * N2T * DD * 2);
    f16*   y1     = (f16*)  alloc((size_t)BB * N1T * DD * 2);
    float* mlog   = (float*)alloc((size_t)BB * N1T * PP * 4);
    float* rmax   = (float*)alloc((size_t)BB * PP * 4);
    float* rsum   = (float*)alloc((size_t)BB * PP * 4);
    float* protos = (float*)alloc((size_t)BB * PP * DD * 4);
    float* qry    = (float*)alloc((size_t)BB * NQ * DD * 4);
    float* qh     = (float*)alloc((size_t)BB * 8 * NQ * 8 * 4);
    float* kp     = (float*)alloc((size_t)BB * N2T * DD * 4);
    float* vp     = (float*)alloc((size_t)BB * N2T * DD * 4);
    float* oatt   = (float*)alloc((size_t)BB * NQ * DD * 4);
    float* x1     = (float*)alloc((size_t)BB * NQ * DD * 4);
    f16* lw1h = (f16*)alloc(64 * 64 * 2);
    f16* lw2h = (f16*)alloc(64 * 64 * 2);
    f16* wkh  = (f16*)alloc(64 * 64 * 2);
    f16* wvh  = (f16*)alloc(64 * 64 * 2);
    f16* pwh  = (f16*)alloc(64 * 64 * 2);
    f16* mwh  = (f16*)alloc(16 * 64 * 2);
    float* bn1s = (float*)alloc(64 * 4);
    float* bn1bv = (float*)alloc(64 * 4);
    float* bn2s = (float*)alloc(64 * 4);
    float* bn2bv = (float*)alloc(64 * 4);

    k_prep_weights<<<1, 256, 0, stream>>>(w_in, pww, mhw, m1lw1, m1lw2,
                                          bn1g, bn1bb, bn1m, bn1v, bn2g, bn2bb, bn2m, bn2v,
                                          lw1h, lw2h, wkh, wvh, pwh, mwh,
                                          bn1s, bn1bv, bn2s, bn2bv);
    k_key<<<(BB * N1T * DD) / 256, 256, 0, stream>>>(memory, pos, key);
    k_query<<<(BB * NQ * DD) / 256, 256, 0, stream>>>(tgt, qpos, qry);
    k_wavelet<<<(BB * N2T * DD) / 256, 256, 0, stream>>>(key, high, low, s_h);
    k_pool_glo<<<BB, 64, 0, stream>>>(high, low, m1gw1, m1gb1, m1gw2, m1gb2, glo1);
    k_mscw_local<<<(BB * N2T) / 16, 32, 0, stream>>>(s_h, lw1h, lw2h, m1lb1, m1lb2, glo1,
                                                     high, low, fre_h);
    k_dwconv<<<(BB * N1T * DD) / 256, 256, 0, stream>>>(key, dww, bn1s, bn1bv, y1);
    k_pointwise<<<(BB * N1T) / 16, 32, 0, stream>>>(y1, pwh, mwh, bn2s, bn2bv, mlog);
    k_softstats<<<BB * PP, 256, 0, stream>>>(mlog, rmax, rsum);
    k_protos<<<BB * PP, 256, 0, stream>>>(mlog, key, rmax, rsum, protos);
    k_qproj<<<BB * NQ, 64, 0, stream>>>(qry, w_in, b_in, qh);
    k_kvproj<<<(BB * N2T) / 16, 32, 0, stream>>>(fre_h, wkh, wvh, b_in, kp, vp);
    k_attn<<<BB * 8, 256, 0, stream>>>(qh, kp, vp, oatt);
    k_outln1<<<BB * NQ, 64, 0, stream>>>(oatt, w_out, b_out, qry, ln1g, ln1b, x1);
    k_final<<<BB, 64, 0, stream>>>(protos, qry, x1, m2lw1, m2lb1, m2lw2, m2lb2,
                                   m2gw1, m2gb1, m2gw2, m2gb2, ln2g, ln2b, out);
}
